// PointNet2SemSegMsg_68212670595318
// MI455X (gfx1250) — matrix operand order repaired and checked by reading, not compiled, not run on hardware
//
#include <hip/hip_runtime.h>
#include <hip/hip_bf16.h>

// ---------------------------------------------------------------------------
// PointNet++ MSG semantic segmentation forward for gfx1250 (MI455X, wave32).
// All MLP/linear layers run through v_wmma_f32_16x16x32_bf16. Activations are
// stored bf16 end-to-end: fragment loads are pure b128 bit-copies (no cvt in
// the K-loop). Each wave computes a 16x32 tile (A reused across two WMMAs).
// ---------------------------------------------------------------------------

#define BN_INV_F 0.9999950000374997f
#define NB 8
#define NPTS 8192
#define NCLS 13

typedef __attribute__((ext_vector_type(16))) __bf16 v16bf;
typedef __attribute__((ext_vector_type(8)))  float  v8f;

__device__ __forceinline__ unsigned short f2bf(float x) {
  union { float f; unsigned u; } c; c.f = x;
  unsigned u = c.u;
  unsigned r = u + 0x7FFFu + ((u >> 16) & 1u);   // round-to-nearest-even
  return (unsigned short)(r >> 16);
}
__device__ __forceinline__ float bf2f(unsigned short b) {
  union { unsigned u; float f; } c; c.u = ((unsigned)b) << 16;
  return c.f;
}

union Frag { v16bf v; unsigned x[8]; };

static inline int rnd32(int v) { return (v + 31) & ~31; }

// ---------------------------------------------------------------------------
// Fused Y = relu((X @ W^T + b) * (g*BN_INV) + bt), all-bf16 storage.
// X: [M, ldX] bf16 zero-padded, ldX % 32 == 0, M % 16 == 0.
// Wbf: [round32(Cout), ldX] bf16 zero-padded.
// Y: [M, ldY] bf16, ldY % 32 == 0 (pad cols written as 0).
// One wave32 per 16x32 tile; grid = (M/16, ldY/32).
// ---------------------------------------------------------------------------
__global__ __launch_bounds__(32) void linear_bn_relu_wmma(
    const unsigned short* __restrict__ X, int ldX,
    const unsigned short* __restrict__ Wbf,
    const float* __restrict__ bias,
    const float* __restrict__ gamma, const float* __restrict__ beta,
    unsigned short* __restrict__ Y, int Cout, int ldY)
{
  const int m0   = blockIdx.x * 16;
  const int n0   = blockIdx.y * 32;
  const int lane = threadIdx.x;
  const int lrow = lane & 15;
  const int hi   = lane >> 4;           // 0: lanes 0-15, 1: lanes 16-31

  const unsigned short* xr  = X   + (size_t)(m0 + lrow) * ldX + hi * 16;
  const unsigned short* wr0 = Wbf + (size_t)(n0 + lrow) * ldX + hi * 16;
  const unsigned short* wr1 = wr0 + (size_t)16 * ldX;

  v8f acc0 = {}, acc1 = {};
  for (int kk = 0; kk < ldX; kk += 32) {
    // A fragment (16x32 bf16): lane half hi -> K {hi*16 + 0..15}
    uint4 a0 = *reinterpret_cast<const uint4*>(xr + kk);
    uint4 a1 = *reinterpret_cast<const uint4*>(xr + kk + 8);
    Frag a;
    a.x[0] = a0.x; a.x[1] = a0.y; a.x[2] = a0.z; a.x[3] = a0.w;
    a.x[4] = a1.x; a.x[5] = a1.y; a.x[6] = a1.z; a.x[7] = a1.w;
    // B fragments: lane half hi -> K {hi*16 + 0..15}
    uint4 q0 = *reinterpret_cast<const uint4*>(wr0 + kk);
    uint4 q1 = *reinterpret_cast<const uint4*>(wr0 + kk + 8);
    uint4 q2 = *reinterpret_cast<const uint4*>(wr1 + kk);
    uint4 q3 = *reinterpret_cast<const uint4*>(wr1 + kk + 8);
    Frag b0, b1;
    b0.x[0] = q0.x; b0.x[1] = q0.y; b0.x[2] = q0.z; b0.x[3] = q0.w;
    b0.x[4] = q1.x; b0.x[5] = q1.y; b0.x[6] = q1.z; b0.x[7] = q1.w;
    b1.x[0] = q2.x; b1.x[1] = q2.y; b1.x[2] = q2.z; b1.x[3] = q2.w;
    b1.x[4] = q3.x; b1.x[5] = q3.y; b1.x[6] = q3.z; b1.x[7] = q3.w;

    acc0 = __builtin_amdgcn_wmma_f32_16x16x32_bf16(
        false, a.v, false, b0.v, (short)0, acc0, false, false);
    acc1 = __builtin_amdgcn_wmma_f32_16x16x32_bf16(
        false, a.v, false, b1.v, (short)0, acc1, false, false);
  }

  unsigned short* yr = Y + (size_t)(m0 + hi * 8) * ldY;
#pragma unroll
  for (int t = 0; t < 2; ++t) {
    int ncol = n0 + t * 16 + lrow;
    float s = 0.f, off = 0.f;
    if (ncol < Cout) {
      s   = gamma[ncol] * BN_INV_F;
      off = bias[ncol] * s + beta[ncol];
    }
    const v8f& acc = t ? acc1 : acc0;
#pragma unroll
    for (int v = 0; v < 8; ++v) {
      unsigned short y = 0;
      if (ncol < Cout) y = f2bf(fmaxf(acc[v] * s + off, 0.f));
      yr[(size_t)v * ldY + ncol] = y;
    }
  }
}

// ---------------------------------------------------------------------------
// Weight f32 [Cout, Cin] -> zero-padded bf16 [rnd32(Cout), ldK].
// ---------------------------------------------------------------------------
__global__ void wconv_kernel(const float* __restrict__ W, int Cout, int Cin,
                             unsigned short* __restrict__ out, int ldK,
                             long long total)
{
  for (long long idx = blockIdx.x * (long long)blockDim.x + threadIdx.x;
       idx < total; idx += (long long)gridDim.x * blockDim.x) {
    int c = (int)(idx % ldK);
    int r = (int)(idx / ldK);
    float v = 0.f;
    if (r < Cout && c < Cin) v = W[(size_t)r * Cin + c];
    out[idx] = f2bf(v);
  }
}

// ---------------------------------------------------------------------------
// features (B,6,N) -> XT0 f32 (B,N,3), PT0 bf16 (B,N,6)
// ---------------------------------------------------------------------------
__global__ void transpose_kernel(const float* __restrict__ f,
                                 float* __restrict__ xt,
                                 unsigned short* __restrict__ pt,
                                 long long total)
{
  for (long long idx = blockIdx.x * (long long)blockDim.x + threadIdx.x;
       idx < total; idx += (long long)gridDim.x * blockDim.x) {
    int n = (int)(idx % NPTS);
    int c = (int)((idx / NPTS) % 6);
    int b = (int)(idx / ((long long)NPTS * 6));
    float v = f[idx];
    pt[((size_t)b * NPTS + n) * 6 + c] = f2bf(v);
    if (c < 3) xt[((size_t)b * NPTS + n) * 3 + c] = v;
  }
}

// ---------------------------------------------------------------------------
// Farthest point sampling: one block per batch, running min-dist in LDS.
// ---------------------------------------------------------------------------
__global__ __launch_bounds__(256) void fps_kernel(
    const float* __restrict__ xyz, int N, int npoint, int* __restrict__ fidx)
{
  __shared__ float sdist[8192];
  __shared__ float rmax[256];
  __shared__ int   ridx[256];
  const int b   = blockIdx.x;
  const int tid = threadIdx.x;
  const float* P = xyz + (size_t)b * N * 3;

  for (int n = tid; n < N; n += 256) sdist[n] = 1e10f;
  __syncthreads();

  int far = 0;
  for (int it = 0; it < npoint; ++it) {
    if (tid == 0) fidx[b * npoint + it] = far;
    float cx = P[far * 3 + 0], cy = P[far * 3 + 1], cz = P[far * 3 + 2];
    float best = -1.f; int bi = 0;
    for (int n = tid; n < N; n += 256) {
      float dx = P[n * 3 + 0] - cx;
      float dy = P[n * 3 + 1] - cy;
      float dz = P[n * 3 + 2] - cz;
      float d  = dx * dx + dy * dy + dz * dz;
      float nd = fminf(sdist[n], d);
      sdist[n] = nd;
      if (nd > best) { best = nd; bi = n; }
    }
    rmax[tid] = best; ridx[tid] = bi;
    __syncthreads();
    for (int s = 128; s > 0; s >>= 1) {
      if (tid < s) {
        if (rmax[tid + s] > rmax[tid] ||
            (rmax[tid + s] == rmax[tid] && ridx[tid + s] < ridx[tid])) {
          rmax[tid] = rmax[tid + s]; ridx[tid] = ridx[tid + s];
        }
      }
      __syncthreads();
    }
    far = ridx[0];
    __syncthreads();
  }
}

__global__ void gather_xyz_kernel(const float* __restrict__ xyz, int Nin,
                                  const int* __restrict__ fidx, int S,
                                  float* __restrict__ nxyz, long long total)
{
  for (long long idx = blockIdx.x * (long long)blockDim.x + threadIdx.x;
       idx < total; idx += (long long)gridDim.x * blockDim.x) {
    int d = (int)(idx % 3);
    int s = (int)((idx / 3) % S);
    int b = (int)(idx / (3LL * S));
    int f = fidx[b * S + s];
    nxyz[idx] = xyz[((size_t)b * Nin + f) * 3 + d];
  }
}

// ---------------------------------------------------------------------------
// Ball query: one wave32 per centroid; first-k in-order via ballot + prefix.
// ---------------------------------------------------------------------------
__global__ __launch_bounds__(32) void query_ball_kernel(
    const float* __restrict__ xyz, int N, const float* __restrict__ nxyz,
    int S, float r2, int k, int* __restrict__ gidx)
{
  const int bs   = blockIdx.x;      // b*S + s
  const int lane = threadIdx.x;
  const int b    = bs / S;
  const float* P = xyz + (size_t)b * N * 3;
  float cx = nxyz[(size_t)bs * 3 + 0];
  float cy = nxyz[(size_t)bs * 3 + 1];
  float cz = nxyz[(size_t)bs * 3 + 2];
  int* out = gidx + (size_t)bs * k;

  int cnt = 0, firstn = N;
  for (int base = 0; base < N && cnt < k; base += 32) {
    int n = base + lane;
    bool ok = false;
    if (n < N) {
      float dx = P[n * 3 + 0] - cx;
      float dy = P[n * 3 + 1] - cy;
      float dz = P[n * 3 + 2] - cz;
      ok = (dx * dx + dy * dy + dz * dz <= r2);
    }
    unsigned msk = (unsigned)__ballot(ok);
    if (ok) {
      int pos = cnt + __popc(msk & ((1u << lane) - 1u));
      if (pos < k) out[pos] = n;
    }
    if (firstn == N && msk) firstn = base + (__ffs((int)msk) - 1);
    cnt += __popc(msk);
  }
  if (firstn >= N) firstn = N - 1;
  for (int p = cnt + lane; p < k; p += 32) out[p] = firstn;
}

// ---------------------------------------------------------------------------
// grouped[b,s,j,:] = concat(pts[gidx] (bf16), xyz[gidx]-new_xyz), pad to Cpad.
// ---------------------------------------------------------------------------
__global__ void group_kernel(const unsigned short* __restrict__ pts, int cin,
                             const float* __restrict__ xyz, int Nin,
                             const float* __restrict__ nxyz,
                             const int* __restrict__ gidx, int S, int k,
                             unsigned short* __restrict__ out, int Cpad,
                             long long total)
{
  for (long long idx = blockIdx.x * (long long)blockDim.x + threadIdx.x;
       idx < total; idx += (long long)gridDim.x * blockDim.x) {
    int c = (int)(idx % Cpad);
    int j = (int)((idx / Cpad) % k);
    int s = (int)((idx / ((long long)Cpad * k)) % S);
    int b = (int)(idx / ((long long)Cpad * k * S));
    int bs = b * S + s;
    unsigned short v = 0;
    if (c < cin + 3) {
      int g = gidx[(size_t)bs * k + j];
      if (c < cin) v = pts[((size_t)b * Nin + g) * cin + c];
      else {
        int d = c - cin;
        v = f2bf(xyz[((size_t)b * Nin + g) * 3 + d] - nxyz[(size_t)bs * 3 + d]);
      }
    }
    out[idx] = v;
  }
}

// bf16 max-pool: monotone bf16<->f32 shift lets us keep the winning bits.
__global__ void maxpool_kernel(const unsigned short* __restrict__ X, int S,
                               int k, int C, unsigned short* __restrict__ P,
                               int Ctot, int coff, long long total)
{
  for (long long idx = blockIdx.x * (long long)blockDim.x + threadIdx.x;
       idx < total; idx += (long long)gridDim.x * blockDim.x) {
    int c  = (int)(idx % C);
    long long bs = idx / C;
    const unsigned short* xr = X + (size_t)bs * k * C + c;
    float m = -3.4e38f; unsigned short mb = 0;
    for (int j = 0; j < k; ++j) {
      unsigned short bbits = xr[(size_t)j * C];
      float f = bf2f(bbits);
      if (f > m) { m = f; mb = bbits; }
    }
    P[(size_t)bs * Ctot + coff + c] = mb;
  }
}

__global__ void copy_skip_kernel(const unsigned short* __restrict__ skip,
                                 int Cs, unsigned short* __restrict__ out,
                                 int Ctot, long long total)
{
  for (long long idx = blockIdx.x * (long long)blockDim.x + threadIdx.x;
       idx < total; idx += (long long)gridDim.x * blockDim.x) {
    int c = (int)(idx % Cs);
    long long i = idx / Cs;
    out[(size_t)i * Ctot + c] = skip[idx];
  }
}

// ---------------------------------------------------------------------------
// 3-NN inverse-squared-distance interpolation (bf16 features).
// ---------------------------------------------------------------------------
__global__ void interp_kernel(const float* __restrict__ xh, int Nh,
                              const float* __restrict__ xl, int Nl,
                              const unsigned short* __restrict__ plo, int C,
                              unsigned short* __restrict__ out, int Ctot,
                              int coff, long long total)
{
  for (long long i = blockIdx.x * (long long)blockDim.x + threadIdx.x;
       i < total; i += (long long)gridDim.x * blockDim.x) {
    int b = (int)(i / Nh);
    float px = xh[(size_t)i * 3 + 0];
    float py = xh[(size_t)i * 3 + 1];
    float pz = xh[(size_t)i * 3 + 2];
    const float* L = xl + (size_t)b * Nl * 3;
    float d0 = 3.4e38f, d1 = 3.4e38f, d2 = 3.4e38f;
    int i0 = 0, i1 = 0, i2 = 0;
    for (int m = 0; m < Nl; ++m) {
      float dx = L[m * 3 + 0] - px;
      float dy = L[m * 3 + 1] - py;
      float dz = L[m * 3 + 2] - pz;
      float d  = dx * dx + dy * dy + dz * dz;
      if (d < d0)      { d2 = d1; i2 = i1; d1 = d0; i1 = i0; d0 = d; i0 = m; }
      else if (d < d1) { d2 = d1; i2 = i1; d1 = d;  i1 = m; }
      else if (d < d2) { d2 = d;  i2 = m; }
    }
    float w0 = 1.f / (d0 + 1e-8f);
    float w1 = 1.f / (d1 + 1e-8f);
    float w2 = 1.f / (d2 + 1e-8f);
    float ws = w0 + w1 + w2;
    w0 /= ws; w1 /= ws; w2 /= ws;
    const unsigned short* r0 = plo + ((size_t)b * Nl + i0) * C;
    const unsigned short* r1 = plo + ((size_t)b * Nl + i1) * C;
    const unsigned short* r2 = plo + ((size_t)b * Nl + i2) * C;
    unsigned short* o = out + (size_t)i * Ctot + coff;
    for (int c = 0; c < C; ++c)
      o[c] = f2bf(w0 * bf2f(r0[c]) + w1 * bf2f(r1[c]) + w2 * bf2f(r2[c]));
  }
}

// ---------------------------------------------------------------------------
// Head classifier: 128 (bf16) -> 13 + softmax (f32 out).
// ---------------------------------------------------------------------------
__global__ void head_softmax_kernel(const unsigned short* __restrict__ H,
                                    const float* __restrict__ W2,
                                    const float* __restrict__ b2,
                                    float* __restrict__ out, long long M)
{
  for (long long i = blockIdx.x * (long long)blockDim.x + threadIdx.x;
       i < M; i += (long long)gridDim.x * blockDim.x) {
    const unsigned short* h = H + (size_t)i * 128;
    float hv[128];
    for (int c = 0; c < 128; ++c) hv[c] = bf2f(h[c]);
    float lg[NCLS];
    float mx = -3.4e38f;
    for (int j = 0; j < NCLS; ++j) {
      float s = b2[j];
      const float* w = W2 + j * 128;
      for (int c = 0; c < 128; ++c) s += hv[c] * w[c];
      lg[j] = s;
      mx = fmaxf(mx, s);
    }
    float se = 0.f;
    for (int j = 0; j < NCLS; ++j) { lg[j] = __expf(lg[j] - mx); se += lg[j]; }
    float inv = 1.f / se;
    for (int j = 0; j < NCLS; ++j) out[(size_t)i * NCLS + j] = lg[j] * inv;
  }
}

// ---------------------------------------------------------------------------
// Host side
// ---------------------------------------------------------------------------
struct LayerP {
  const float *W, *b, *g, *bt;
  int cin, cout;
  unsigned short* Wbf;   // padded bf16 weights [rnd32(cout), rnd32(cin)]
};

static inline unsigned gsz(long long total) {
  long long g = (total + 255) / 256;
  if (g < 1) g = 1;
  if (g > 262144) g = 262144;
  return (unsigned)g;
}

extern "C" void kernel_launch(void* const* d_in, const int* in_sizes, int n_in,
                              void* d_out, int out_size, void* d_ws, size_t ws_size,
                              hipStream_t stream)
{
  (void)in_sizes; (void)n_in; (void)out_size; (void)ws_size;

  // ---- workspace bump allocator -------------------------------------------
  char* base = (char*)d_ws;
  size_t off = 0;
  auto allocF = [&](size_t n) {
    float* p = (float*)(base + off);
    off = (off + n * sizeof(float) + 255) & ~(size_t)255;
    return p;
  };
  auto allocI = [&](size_t n) {
    int* p = (int*)(base + off);
    off = (off + n * sizeof(int) + 255) & ~(size_t)255;
    return p;
  };
  auto allocU = [&](size_t n) {
    unsigned short* p = (unsigned short*)(base + off);
    off = (off + n * sizeof(unsigned short) + 255) & ~(size_t)255;
    return p;
  };

  // ---- parameter unpack (jax pytree order: sorted dict keys: fp, head, sa) --
  int ai = 0;
  const float* feats = (const float*)d_in[ai++];
  auto nextL = [&](int ci, int co) {
    LayerP L;
    L.W  = (const float*)d_in[ai + 0];
    L.b  = (const float*)d_in[ai + 1];
    L.g  = (const float*)d_in[ai + 2];
    L.bt = (const float*)d_in[ai + 3];
    L.cin = ci; L.cout = co; ai += 4;
    int rows = rnd32(co), ldK = rnd32(ci);
    L.Wbf = allocU((size_t)rows * ldK);
    long long tot = (long long)rows * ldK;
    wconv_kernel<<<gsz(tot), 256, 0, stream>>>(L.W, co, ci, L.Wbf, ldK, tot);
    return L;
  };

  LayerP fpL[4][3]; int fpCnt[4] = {2, 2, 2, 3};
  { int w[4][4] = {{1536,256,256,0},{512,256,256,0},{352,256,128,0},{128,128,128,128}};
    for (int t = 0; t < 4; ++t)
      for (int i = 0; i < fpCnt[t]; ++i)
        fpL[t][i] = nextL(w[t][i], w[t][i + 1]); }

  LayerP headc1 = nextL(128, 128);
  const float* headW2 = (const float*)d_in[ai++];
  const float* headb2 = (const float*)d_in[ai++];

  LayerP saL[4][2][3];
  { int w[4][2][4] = {
      {{9,16,16,32},   {9,32,32,64}},
      {{99,64,64,128}, {99,64,96,128}},
      {{259,128,196,256},{259,128,196,256}},
      {{515,256,256,512},{515,256,384,512}}};
    for (int l = 0; l < 4; ++l)
      for (int j = 0; j < 2; ++j)
        for (int t = 0; t < 3; ++t)
          saL[l][j][t] = nextL(w[l][j][t], w[l][j][t + 1]); }

  // ---- activation / index buffers -----------------------------------------
  float* XT0 = allocF((size_t)NB * NPTS * 3);
  unsigned short* PT0 = allocU((size_t)NB * NPTS * 6);
  float* XYZ1 = allocF((size_t)NB * 1024 * 3);
  float* XYZ2 = allocF((size_t)NB * 256 * 3);
  float* XYZ3 = allocF((size_t)NB * 64 * 3);
  float* XYZ4 = allocF((size_t)NB * 16 * 3);
  unsigned short* P1 = allocU((size_t)NB * 1024 * 96);
  unsigned short* P2 = allocU((size_t)NB * 256 * 256);
  unsigned short* P3 = allocU((size_t)NB * 64 * 512);
  unsigned short* P4 = allocU((size_t)NB * 16 * 1024);
  unsigned short* F3 = allocU((size_t)NB * 64 * 256);
  unsigned short* F2 = allocU((size_t)NB * 256 * 256);
  unsigned short* F1 = allocU((size_t)NB * 1024 * 128);
  unsigned short* F0 = allocU((size_t)NB * NPTS * 128);
  unsigned short* HH = allocU((size_t)NB * NPTS * 128);
  unsigned short* GRP = allocU((size_t)NB * 1024 * 32 * 32);
  unsigned short* T1 = allocU((size_t)NB * 1024 * 32 * 64);
  unsigned short* T2 = allocU((size_t)NB * 1024 * 32 * 64);
  int* FIDX = allocI((size_t)NB * 1024);
  int* GIDX = allocI((size_t)NB * 1024 * 32);

  auto linear = [&](const unsigned short* Xp, long long M, const LayerP& L,
                    unsigned short* Yp) {
    int ldX = rnd32(L.cin), ldY = rnd32(L.cout);
    dim3 g((unsigned)(M / 16), (unsigned)(ldY / 32));
    linear_bn_relu_wmma<<<g, dim3(32), 0, stream>>>(
        Xp, ldX, L.Wbf, L.b, L.g, L.bt, Yp, L.cout, ldY);
  };

  // ---- input transpose -----------------------------------------------------
  {
    long long tot = (long long)NB * 6 * NPTS;
    transpose_kernel<<<gsz(tot), 256, 0, stream>>>(feats, XT0, PT0, tot);
  }

  // ---- SA levels -----------------------------------------------------------
  float* XYZL[5] = {XT0, XYZ1, XYZ2, XYZ3, XYZ4};
  unsigned short* PTSL[5] = {PT0, P1, P2, P3, P4};
  int NPT[5] = {NPTS, 1024, 256, 64, 16};
  int CCH[5] = {6, 96, 256, 512, 1024};
  float radii[4][2] = {{0.05f, 0.1f}, {0.1f, 0.2f}, {0.2f, 0.4f}, {0.4f, 0.8f}};
  int ksz[2] = {16, 32};

  for (int l = 0; l < 4; ++l) {
    int nin = NPT[l], nou = NPT[l + 1], ci = CCH[l];
    fps_kernel<<<NB, 256, 0, stream>>>(XYZL[l], nin, nou, FIDX);
    { long long tot = (long long)NB * nou * 3;
      gather_xyz_kernel<<<gsz(tot), 256, 0, stream>>>(XYZL[l], nin, FIDX, nou,
                                                      XYZL[l + 1], tot); }
    int coff = 0;
    for (int j = 0; j < 2; ++j) {
      int k = ksz[j];
      float r = radii[l][j];
      query_ball_kernel<<<NB * nou, 32, 0, stream>>>(
          XYZL[l], nin, XYZL[l + 1], nou, r * r, k, GIDX);
      long long Mrows = (long long)NB * nou * k;
      int Cpad = rnd32(ci + 3);
      { long long tot = Mrows * Cpad;
        group_kernel<<<gsz(tot), 256, 0, stream>>>(
            PTSL[l], ci, XYZL[l], nin, XYZL[l + 1], GIDX, nou, k, GRP, Cpad, tot); }
      const unsigned short* cur = GRP;
      for (int t = 0; t < 3; ++t) {
        unsigned short* dst = (t % 2 == 0) ? T1 : T2;
        linear(cur, Mrows, saL[l][j][t], dst);
        cur = dst;
      }
      int cob = saL[l][j][2].cout;   // always a multiple of 32
      { long long tot = (long long)NB * nou * cob;
        maxpool_kernel<<<gsz(tot), 256, 0, stream>>>(
            cur, nou, k, cob, PTSL[l + 1], CCH[l + 1], coff, tot); }
      coff += cob;
    }
  }

  // ---- FP levels -----------------------------------------------------------
  struct FpCfg { int hi, lo; const unsigned short* skip; int skipC;
                 const unsigned short* plo; int ploC; unsigned short* outF; };
  FpCfg fpc[4] = {
      {3, 4, P3, 512, P4, 1024, F3},
      {2, 3, P2, 256, F3, 256, F2},
      {1, 2, P1, 96,  F2, 256, F1},
      {0, 1, nullptr, 0, F1, 128, F0}};
  for (int t = 0; t < 4; ++t) {
    int Nh = NPT[fpc[t].hi], Nl = NPT[fpc[t].lo];
    int Ctot = fpc[t].skipC + fpc[t].ploC;   // 1536 / 512 / 352 / 128: all %32==0
    long long M = (long long)NB * Nh;
    if (fpc[t].skip) {
      long long tot = M * fpc[t].skipC;
      copy_skip_kernel<<<gsz(tot), 256, 0, stream>>>(fpc[t].skip, fpc[t].skipC,
                                                     T2, Ctot, tot);
    }
    interp_kernel<<<gsz(M), 256, 0, stream>>>(
        XYZL[fpc[t].hi], Nh, XYZL[fpc[t].lo], Nl, fpc[t].plo, fpc[t].ploC,
        T2, Ctot, fpc[t].skipC, M);
    const unsigned short* cur = T2;
    for (int i = 0; i < fpCnt[t]; ++i) {
      unsigned short* dst =
          (i == fpCnt[t] - 1) ? fpc[t].outF : ((i % 2 == 0) ? T1 : T2);
      linear(cur, M, fpL[t][i], dst);
      cur = dst;
    }
  }

  // ---- head ----------------------------------------------------------------
  long long Mpts = (long long)NB * NPTS;
  linear(F0, Mpts, headc1, HH);
  head_softmax_kernel<<<gsz(Mpts), 256, 0, stream>>>(HH, headW2, headb2,
                                                     (float*)d_out, Mpts);
}
